// TopoMLP_69913477644879
// MI455X (gfx1250) — compile-verified, matrix operand order checked by reading, not run on hardware
//
#include <hip/hip_runtime.h>
#include <hip/hip_bf16.h>

typedef __attribute__((ext_vector_type(16))) _Float16 v16h;
typedef __attribute__((ext_vector_type(8)))  float    v8f;

#define LDX 136   // padded ld (halves) for 128-wide rows: 272 B/row, bank stride 4 DW, 16B aligned
#define LDW 264   // padded ld (halves) for 256-wide rows: 528 B/row, bank stride 4 DW, 16B aligned

// Unified A/B fragment loader (symmetric lane layouts): 16x32 f16 operand from
// row-major [row][K] LDS data.
//   row (A: M, B: N) = lane & 15
//   element j -> K = (lane>=16 ? 8 : 0) + (j & 7) + (j >= 8 ? 16 : 0)
// K-contiguous => two 16B ds_load_b128 per fragment.
__device__ __forceinline__ v16h ldsFrag(const _Float16* __restrict__ base, int ld,
                                        int R0, int K0, int lane) {
  const int hi = lane >> 4;
  const int r  = lane & 15;
  const _Float16* p = base + (R0 + r) * ld + K0 + (hi << 3);
  v16h f;
#pragma unroll
  for (int j = 0; j < 16; ++j) f[j] = p[(j & 7) + ((j & 8) << 1)];
  return f;
}

// Fast sigmoid: v_mul + v_exp_f32 + v_add + v_rcp_f32 (no IEEE div expansion).
// v_rcp_f32 is ~1 ulp, far below the f16 quantization that follows.
__device__ __forceinline__ float sigmoidf(float v) {
  return __builtin_amdgcn_rcpf(1.0f + __expf(-v));
}

// One workgroup: sub-network i = blockIdx.y, rows [blockIdx.x*128, +128).
// 256 threads = 8 waves arranged 4(M) x 2(N):
//   wave w: M-stripe [32*(w&3), +32), N-half (w>>2).
__global__ __launch_bounds__(256)
void topomlp_fused_kernel(const float* __restrict__ x,
                          const float* __restrict__ W0,
                          const float* __restrict__ W1,
                          const float* __restrict__ W2,
                          float* __restrict__ out) {
  extern __shared__ _Float16 smem[];
  _Float16* sX  = smem;                 // 128 x LDX : X tile (layer0 A); reused as H2
  _Float16* sH  = sX + 128 * LDX;       // 128 x LDW : layer-0 activations (layer1 A)
  _Float16* sW  = sH + 128 * LDW;       // max(256*LDX, 128*LDW): W0[i]^T, then W1[i]
  _Float16* sW2 = sW + 256 * LDX;       // 128

  const int tid  = threadIdx.x;
  const int wave = tid >> 5;
  const int lane = tid & 31;
  const int hi   = lane >> 4;
  const int nl   = lane & 15;
  const int mw   = wave & 3;            // M stripe: rows [32*mw, +32)
  const int nw   = wave >> 1 >> 1;      // N half: 0 or 1
  const int i    = blockIdx.y;
  const int n0   = blockIdx.x * 128;

  // ---- cooperative loads (f32 -> f16) ----
  // X tile (128 x 128), row-major, coalesced float4.
  const float4* xp = (const float4*)(x + (size_t)n0 * 128);
  for (int f = tid; f < 128 * 32; f += 256) {
    const int r = f >> 5, c = (f & 31) << 2;
    const float4 v = xp[f];
    _Float16* d = sX + r * LDX + c;
    d[0] = (_Float16)v.x; d[1] = (_Float16)v.y; d[2] = (_Float16)v.z; d[3] = (_Float16)v.w;
  }
  // W0[i] (128 k-rows x 256 n-cols) -> TRANSPOSED into sW as [n][k] (256 x LDX),
  // 4k x 4n blocks: 4 coalesced float4 loads -> 4 k-contiguous b64 stores.
  const float4* w0p = (const float4*)(W0 + (size_t)i * 128 * 256);
  for (int idx = tid; idx < 2048; idx += 256) {
    const int bk = idx >> 6;        // 0..31 : k = 4*bk
    const int bn = idx & 63;        // 0..63 : n = 4*bn
    const float4 r0 = w0p[(4 * bk + 0) * 64 + bn];
    const float4 r1 = w0p[(4 * bk + 1) * 64 + bn];
    const float4 r2 = w0p[(4 * bk + 2) * 64 + bn];
    const float4 r3 = w0p[(4 * bk + 3) * 64 + bn];
    _Float16* d0 = sW + (4 * bn + 0) * LDX + 4 * bk;
    d0[0] = (_Float16)r0.x; d0[1] = (_Float16)r1.x; d0[2] = (_Float16)r2.x; d0[3] = (_Float16)r3.x;
    _Float16* d1 = sW + (4 * bn + 1) * LDX + 4 * bk;
    d1[0] = (_Float16)r0.y; d1[1] = (_Float16)r1.y; d1[2] = (_Float16)r2.y; d1[3] = (_Float16)r3.y;
    _Float16* d2 = sW + (4 * bn + 2) * LDX + 4 * bk;
    d2[0] = (_Float16)r0.z; d2[1] = (_Float16)r1.z; d2[2] = (_Float16)r2.z; d2[3] = (_Float16)r3.z;
    _Float16* d3 = sW + (4 * bn + 3) * LDX + 4 * bk;
    d3[0] = (_Float16)r0.w; d3[1] = (_Float16)r1.w; d3[2] = (_Float16)r2.w; d3[3] = (_Float16)r3.w;
  }
  if (tid < 128) sW2[tid] = (_Float16)W2[(size_t)i * 128 + tid];
  __syncthreads();

  // ---- layer 0: H = sigmoid(X @ W0[i])  (128x256), K = 128 ----
  // wave tile: 32(M) x 128(N); each B fragment feeds 2 WMMAs.
  v16h a0[2][4];
#pragma unroll
  for (int mt = 0; mt < 2; ++mt)
#pragma unroll
    for (int kt = 0; kt < 4; ++kt)
      a0[mt][kt] = ldsFrag(sX, LDX, mw * 32 + mt * 16, kt * 32, lane);

#pragma unroll
  for (int nt = 0; nt < 8; ++nt) {
    const int nb = nw * 128 + nt * 16;
    v8f acc0 = {}, acc1 = {};
#pragma unroll
    for (int kt = 0; kt < 4; ++kt) {
      v16h b = ldsFrag(sW, LDX, nb, kt * 32, lane);
      acc0 = __builtin_amdgcn_wmma_f32_16x16x32_f16(false, a0[0][kt], false, b, (short)0, acc0, false, false);
      acc1 = __builtin_amdgcn_wmma_f32_16x16x32_f16(false, a0[1][kt], false, b, (short)0, acc1, false, false);
    }
    _Float16* d0 = sH + (mw * 32 + hi * 8) * LDW + nb + nl;
#pragma unroll
    for (int r = 0; r < 8; ++r) {
      d0[r * LDW]        = (_Float16)sigmoidf(acc0[r]);
      d0[(16 + r) * LDW] = (_Float16)sigmoidf(acc1[r]);
    }
  }
  __syncthreads();  // all waves done reading sW (W0^T) and writing sH

  // ---- swap in W1[i] (128 o-rows x 256 k-cols, row-major: k contiguous) ----
  const float4* w1p = (const float4*)(W1 + (size_t)i * 128 * 256);
  for (int f = tid; f < 128 * 64; f += 256) {
    const int r = f >> 6, c = (f & 63) << 2;
    const float4 v = w1p[f];
    _Float16* d = sW + r * LDW + c;
    d[0] = (_Float16)v.x; d[1] = (_Float16)v.y; d[2] = (_Float16)v.z; d[3] = (_Float16)v.w;
  }
  __syncthreads();

  // ---- layer 1: H2 = sigmoid(H @ W1[i]^T)  (128x128), K = 256 ----
  // wave tile: 32(M) x 64(N).
  v16h a1[2][8];
#pragma unroll
  for (int mt = 0; mt < 2; ++mt)
#pragma unroll
    for (int kt = 0; kt < 8; ++kt)
      a1[mt][kt] = ldsFrag(sH, LDW, mw * 32 + mt * 16, kt * 32, lane);

  _Float16* sH2 = sX;  // reuse X buffer
#pragma unroll
  for (int nt = 0; nt < 4; ++nt) {
    const int nb = nw * 64 + nt * 16;
    v8f acc0 = {}, acc1 = {};
#pragma unroll
    for (int kt = 0; kt < 8; ++kt) {
      // B[k][n] = W1[i][n][k] = sW[n*LDW + k]  (row n, contiguous k)
      v16h b = ldsFrag(sW, LDW, nb, kt * 32, lane);
      acc0 = __builtin_amdgcn_wmma_f32_16x16x32_f16(false, a1[0][kt], false, b, (short)0, acc0, false, false);
      acc1 = __builtin_amdgcn_wmma_f32_16x16x32_f16(false, a1[1][kt], false, b, (short)0, acc1, false, false);
    }
    _Float16* d0 = sH2 + (mw * 32 + hi * 8) * LDX + nb + nl;
#pragma unroll
    for (int r = 0; r < 8; ++r) {
      d0[r * LDX]        = (_Float16)sigmoidf(acc0[r]);
      d0[(16 + r) * LDX] = (_Float16)sigmoidf(acc1[r]);
    }
  }
  __syncthreads();

  // ---- layer 2: out[n0+r, i] = H2[r,:] . W2[i,:]  (f32 accumulate) ----
  if (tid < 128) {
    const _Float16* row = sH2 + tid * LDX;
    float sum = 0.0f;
#pragma unroll 16
    for (int k = 0; k < 128; ++k) sum += (float)row[k] * (float)sW2[k];
    out[(size_t)(n0 + tid) * 128 + i] = sum;
  }
}

extern "C" void kernel_launch(void* const* d_in, const int* in_sizes, int n_in,
                              void* d_out, int out_size, void* d_ws, size_t ws_size,
                              hipStream_t stream) {
  const float* x  = (const float*)d_in[0];   // (N, 128)
  const float* W0 = (const float*)d_in[1];   // (128, 128, 256)
  const float* W1 = (const float*)d_in[2];   // (128, 128, 256)
  const float* W2 = (const float*)d_in[3];   // (128, 1, 128)
  float* out = (float*)d_out;                // (N, 128)

  const int N = in_sizes[0] / 128;           // 4096
  const size_t lds_bytes =
      (size_t)(128 * LDX + 128 * LDW + 256 * LDX + 128) * sizeof(_Float16);  // ~172 KB

  dim3 grid(N / 128, 128);
  dim3 block(256);
  hipLaunchKernelGGL(topomlp_fused_kernel, grid, block, lds_bytes, stream,
                     x, W0, W1, W2, out);
}
